// YakirCorrelation_55886114455785
// MI455X (gfx1250) — matrix operand order repaired and checked by reading, not compiled, not run on hardware
//
#include <hip/hip_runtime.h>
#include <hip/hip_bf16.h>

typedef __attribute__((ext_vector_type(16))) _Float16 v16h;
typedef __attribute__((ext_vector_type(8)))  _Float16 v8h;
typedef __attribute__((ext_vector_type(8)))  float    v8f;
typedef __attribute__((ext_vector_type(4)))  float    v4f;

#define BATCH 4
#define NSEQ  1024
#define CKCH  512
#define FS    7
#define PADW  3

// ---------------------------------------------------------------------------
// Weight convert f32 (co,ci,t) -> f16 (t,co,ci)  (one NB slice per launch)
// ---------------------------------------------------------------------------
__global__ void yk_wcvt(const float* __restrict__ src, _Float16* __restrict__ dst, int total) {
    int idx = blockIdx.x * blockDim.x + threadIdx.x;
    if (idx >= total) return;
    int ci = idx % CKCH;
    int r  = idx / CKCH;
    int co = r % CKCH;
    int t  = r / CKCH;
    dst[idx] = (_Float16)src[((size_t)co * CKCH + ci) * FS + t];
}

__global__ void yk_cvt(const float* __restrict__ src, _Float16* __restrict__ dst, int total) {
    int idx = blockIdx.x * blockDim.x + threadIdx.x;
    if (idx >= total) return;
    dst[idx] = (_Float16)src[idx];
}

__global__ void yk_zero(float* s) { if (threadIdx.x < 8) s[threadIdx.x] = 0.f; }

// ---------------------------------------------------------------------------
// WMMA conv1d with LDS-staged input:
//   out[co,x] = sum_t sum_ci W[t,co,ci] * in[ci, stride*x + t - 3]
// in: (B,512,Lin) ch-major f32.  wt: (7,512,512) f16.  out: (B,512,Lout) f32.
// Block = 256 threads (8 waves) computes 128(co) x 32(x) outputs.
// Stage: whole input patch (span = stride*32+6 positions x 512 ci) as f16 in
// LDS, layout [pos][ci] so each B fragment is two ds_load_b128 of contiguous
// f16.  Padding / stride / rep2 upsampling folded into staging.
// Each wave: 16co x 32x = two D tiles sharing the A fragment (2 wmma/K-step).
// grid: x = ceil(Lout/32), y = 4, z = B;  dynamic LDS = span*512*2 bytes.
// ---------------------------------------------------------------------------
__global__ void yk_conv_wmma(const float* __restrict__ in, const _Float16* __restrict__ wt,
                             const float* __restrict__ bias, float* __restrict__ out,
                             int Lin, int Lout, int stride, int rep2, int accum) {
    extern __shared__ _Float16 sIn[];          // [span][512]
    const int lane  = threadIdx.x & 31;
    const int wv    = threadIdx.x >> 5;
    const int bz    = blockIdx.z;
    const int xbase = blockIdx.x * 32;
    const int co0   = blockIdx.y * 128 + wv * 16;
    const int arow  = lane & 15;               // A: row in tile
    const int akb   = (lane < 16) ? 0 : 16;    // A: K base (contiguous, matches B)
    const int bkb   = (lane < 16) ? 0 : 16;    // B: K base (contiguous, matches A)
    const int lx0   = lane & 15;               // local x of subtile 0
    const size_t inB  = (size_t)bz * CKCH * Lin;
    const size_t outB = (size_t)bz * CKCH * Lout;

    // ---- stage input patch to LDS (transposed, f16, zero-padded) ----
    const int span  = stride * 32 + 6;
    const int espan = rep2 ? (2 * Lin) : Lin;
    const int g0    = stride * xbase - PADW;
    for (int idx = threadIdx.x; idx < CKCH * span; idx += 256) {
        int ci = idx / span;
        int p  = idx - ci * span;
        int g  = g0 + p;
        float v = 0.f;
        if (g >= 0 && g < espan) v = in[inB + (size_t)ci * Lin + (rep2 ? (g >> 1) : g)];
        sIn[(size_t)p * CKCH + ci] = (_Float16)v;
    }
    __syncthreads();

    v8f c0 = {};
    v8f c1 = {};
    for (int t = 0; t < FS; ++t) {
        const _Float16* sB0 = sIn + (size_t)(stride * lx0 + t) * CKCH + bkb;
        const _Float16* sB1 = sIn + (size_t)(stride * (lx0 + 16) + t) * CKCH + bkb;
        const _Float16* wrow = wt + ((size_t)(t * CKCH + (co0 + arow))) * CKCH;
#pragma unroll 4
        for (int kc = 0; kc < CKCH / 32; ++kc) {
            // A fragment: two adjacent 8xf16 runs, contiguous k = 16h + s (as B)
            const _Float16* wp = wrow + kc * 32;
            __builtin_prefetch(wp + CKCH, 0, 1);
            v8h a0 = *(const v8h*)(wp + akb);
            v8h a1 = *(const v8h*)(wp + 8 + akb);
            v16h a;
#pragma unroll
            for (int j = 0; j < 8; ++j) { a[j] = a0[j]; a[j + 8] = a1[j]; }
            // B fragments: 16 contiguous f16 from LDS (two b128 reads each)
            const v8h* bp0 = (const v8h*)(sB0 + kc * 32);
            v8h b0lo = bp0[0], b0hi = bp0[1];
            v16h b0;
#pragma unroll
            for (int j = 0; j < 8; ++j) { b0[j] = b0lo[j]; b0[j + 8] = b0hi[j]; }
            c0 = __builtin_amdgcn_wmma_f32_16x16x32_f16(false, a, false, b0, (short)0, c0,
                                                        false, false);
            const v8h* bp1 = (const v8h*)(sB1 + kc * 32);
            v8h b1lo = bp1[0], b1hi = bp1[1];
            v16h b1;
#pragma unroll
            for (int j = 0; j < 8; ++j) { b1[j] = b1lo[j]; b1[j + 8] = b1hi[j]; }
            c1 = __builtin_amdgcn_wmma_f32_16x16x32_f16(false, a, false, b1, (short)0, c1,
                                                        false, false);
        }
    }

    // ---- store both D subtiles ----
    const int x0 = xbase + lx0;
    const int x1 = x0 + 16;
#pragma unroll
    for (int j = 0; j < 8; ++j) {
        int r = j + ((lane < 16) ? 0 : 8);
        float bv = bias[co0 + r];
        if (x0 < Lout) {
            size_t idx = outB + (size_t)(co0 + r) * Lout + x0;
            float val = c0[j] + bv;
            out[idx] = accum ? (out[idx] + val) : val;
        }
        if (x1 < Lout) {
            size_t idx = outB + (size_t)(co0 + r) * Lout + x1;
            float val = c1[j] + bv;
            out[idx] = accum ? (out[idx] + val) : val;
        }
    }
}

// ---------------------------------------------------------------------------
// WMMA fully-connected: out = X @ W^T + b.  X: (B, N, K) row-major f32.
// W: (M, K) f16.  outChMajor=1 -> out[b][m*N+n] (ch-major), else out[b][n*M+m].
// reluB/scaleB applied to X elements on load.  B fragment K-run is contiguous
// in X rows -> four b128 loads + convert.
// grid: x = N/16, y = M/128, z = B; block = 256
// ---------------------------------------------------------------------------
__global__ void yk_fc_wmma(const float* __restrict__ X, const _Float16* __restrict__ W,
                           const float* __restrict__ bias, float* __restrict__ out,
                           int N, int K, int M, int outChMajor, float scaleB, int reluB) {
    const int lane = threadIdx.x & 31;
    const int wv   = threadIdx.x >> 5;
    const int bz   = blockIdx.z;
    const int x0   = blockIdx.x * 16;
    const int m0   = blockIdx.y * 128 + wv * 16;
    const int arow = lane & 15;
    const int akb  = (lane < 16) ? 0 : 16;
    const int bkb  = (lane < 16) ? 0 : 16;
    const int x    = x0 + (lane & 15);
    const float* xrow = X + (size_t)bz * N * K + (size_t)x * K;

    v8f c = {};
    for (int kc = 0; kc < K / 32; ++kc) {
        const _Float16* wp = W + (size_t)(m0 + arow) * K + kc * 32;
        v8h a0 = *(const v8h*)(wp + akb);
        v8h a1 = *(const v8h*)(wp + 8 + akb);
        v16h a;
#pragma unroll
        for (int j = 0; j < 8; ++j) { a[j] = a0[j]; a[j + 8] = a1[j]; }
        const float* xp = xrow + kc * 32 + bkb;
        v4f f0 = *(const v4f*)(xp + 0);
        v4f f1 = *(const v4f*)(xp + 4);
        v4f f2 = *(const v4f*)(xp + 8);
        v4f f3 = *(const v4f*)(xp + 12);
        float fl[16];
#pragma unroll
        for (int j = 0; j < 4; ++j) {
            fl[j] = f0[j]; fl[4 + j] = f1[j]; fl[8 + j] = f2[j]; fl[12 + j] = f3[j];
        }
        v16h bf;
#pragma unroll
        for (int j = 0; j < 16; ++j) {
            float v = fl[j];
            if (reluB) v = v > 0.f ? v : 0.f;
            bf[j] = (_Float16)(v * scaleB);
        }
        c = __builtin_amdgcn_wmma_f32_16x16x32_f16(false, a, false, bf, (short)0, c,
                                                   false, false);
    }
#pragma unroll
    for (int j = 0; j < 8; ++j) {
        int r = j + ((lane < 16) ? 0 : 8);
        float val = c[j] + bias[m0 + r];
        if (outChMajor)
            out[(size_t)bz * M * N + (size_t)(m0 + r) * N + x] = val;
        else
            out[(size_t)bz * N * M + (size_t)x * M + (m0 + r)] = val;
    }
}

// ---------------------------------------------------------------------------
// Forward sparse DFT: input flat viewed z[j,i] = flat[i*512+j], i in [0,L)
// X[b][m][j] = sum_i z[j,i] * e^{-2pi i m i / L},  m < l
// ---------------------------------------------------------------------------
__global__ void yk_dft(const float* __restrict__ flat, float* __restrict__ Xre,
                       float* __restrict__ Xim, int L, int l) {
    int tid = blockIdx.x * blockDim.x + threadIdx.x;
    int total = BATCH * CKCH * l;
    if (tid >= total) return;
    int j = tid % CKCH;
    int m = (tid / CKCH) % l;
    int b = tid / (CKCH * l);
    const float* f = flat + (size_t)b * CKCH * L;
    float re = 0.f, im = 0.f;
    float w = -6.28318530717958647692f * (float)m / (float)L;
    for (int i = 0; i < L; ++i) {
        float v = f[(size_t)i * CKCH + j];
        float s, ct;
        __sincosf(w * (float)i, &s, &ct);
        re += v * ct;
        im += v * s;
    }
    Xre[((size_t)b * 16 + m) * CKCH + j] = re;
    Xim[((size_t)b * 16 + m) * CKCH + j] = im;
}

// Complex mode mix: G[b][m][o] = sum_j X[b][m][j] * (wr + i*wi)[j][o][m]
__global__ void yk_mix(const float* __restrict__ Xre, const float* __restrict__ Xim,
                       const float* __restrict__ wr, const float* __restrict__ wi,
                       float* __restrict__ Gre, float* __restrict__ Gim, int l) {
    int tid = blockIdx.x * blockDim.x + threadIdx.x;
    int total = BATCH * CKCH * l;
    if (tid >= total) return;
    int o = tid % CKCH;
    int m = (tid / CKCH) % l;
    int b = tid / (CKCH * l);
    const float* xr = Xre + ((size_t)b * 16 + m) * CKCH;
    const float* xi = Xim + ((size_t)b * 16 + m) * CKCH;
    float re = 0.f, im = 0.f;
    for (int j = 0; j < CKCH; ++j) {
        float ar = xr[j], ai = xi[j];
        size_t wI = ((size_t)j * CKCH + o) * 16 + m;
        float br = wr[wI], bi = wi[wI];
        re += ar * br - ai * bi;
        im += ar * bi + ai * br;
    }
    Gre[((size_t)b * 16 + m) * CKCH + o] = re;
    Gim[((size_t)b * 16 + m) * CKCH + o] = im;
}

// Inverse DFT of 16-mode spectrum + fused global sum/sumsq stats.
__global__ void yk_idft(const float* __restrict__ Gre, const float* __restrict__ Gim,
                        float* __restrict__ outf, float* __restrict__ stats, int L, int l) {
    __shared__ float ssum[256];
    __shared__ float ssq[256];
    long long tid = (long long)blockIdx.x * blockDim.x + threadIdx.x;
    long long total = (long long)BATCH * CKCH * L;
    float acc = 0.f;
    bool ok = tid < total;
    if (ok) {
        int o = (int)(tid % CKCH);
        int i = (int)((tid / CKCH) % L);
        int b = (int)(tid / ((long long)CKCH * L));
        float w = 6.28318530717958647692f * (float)i / (float)L;
        for (int m = 0; m < l; ++m) {
            float gr = Gre[((size_t)b * 16 + m) * CKCH + o];
            float gi = Gim[((size_t)b * 16 + m) * CKCH + o];
            float s, ct;
            __sincosf(w * (float)m, &s, &ct);
            float fac = (m == 0 || 2 * m == L) ? 1.f : 2.f;
            acc += fac * (gr * ct - gi * s);
        }
        acc /= (float)L;
        outf[tid] = acc;
    }
    ssum[threadIdx.x] = ok ? acc : 0.f;
    ssq[threadIdx.x]  = ok ? acc * acc : 0.f;
    __syncthreads();
    for (int s = 128; s > 0; s >>= 1) {
        if (threadIdx.x < s) {
            ssum[threadIdx.x] += ssum[threadIdx.x + s];
            ssq[threadIdx.x]  += ssq[threadIdx.x + s];
        }
        __syncthreads();
    }
    if (threadIdx.x == 0) {
        atomicAdd(&stats[0], ssum[0]);
        atomicAdd(&stats[1], ssq[0]);
    }
}

// Standalone sum/sumsq reduction (for reconstruction normalize)
__global__ void yk_stats(const float* __restrict__ buf, float* __restrict__ stats,
                         long long n) {
    __shared__ float ssum[256];
    __shared__ float ssq[256];
    long long tid = (long long)blockIdx.x * blockDim.x + threadIdx.x;
    float v = (tid < n) ? buf[tid] : 0.f;
    ssum[threadIdx.x] = v;
    ssq[threadIdx.x]  = v * v;
    __syncthreads();
    for (int s = 128; s > 0; s >>= 1) {
        if (threadIdx.x < s) {
            ssum[threadIdx.x] += ssum[threadIdx.x + s];
            ssq[threadIdx.x]  += ssq[threadIdx.x + s];
        }
        __syncthreads();
    }
    if (threadIdx.x == 0) {
        atomicAdd(&stats[0], ssum[0]);
        atomicAdd(&stats[1], ssq[0]);
    }
}

// Normalize: v = (v - mean)*rstd  (subMean) or v*rstd, std with ddof=1
__global__ void yk_norm(float* __restrict__ buf, const float* __restrict__ stats,
                        long long n, int subMean) {
    long long tid = (long long)blockIdx.x * blockDim.x + threadIdx.x;
    if (tid >= n) return;
    float fn = (float)n;
    float mean = stats[0] / fn;
    float var  = (stats[1] - fn * mean * mean) / (fn - 1.f);
    float rstd = rsqrtf(var);
    float v = buf[tid];
    buf[tid] = subMean ? (v - mean) * rstd : v * rstd;
}

// Between blocks: true 512xN transpose with fused scale (1/sqrt(C)) + relu
__global__ void yk_relu_tr(const float* __restrict__ in, float* __restrict__ out,
                           float scale) {
    int tid = blockIdx.x * blockDim.x + threadIdx.x;   // total 4*512*1024
    int n  = tid % NSEQ;
    int ch = (tid / NSEQ) % CKCH;
    int b  = tid / (NSEQ * CKCH);
    float v = in[(size_t)b * CKCH * NSEQ + (size_t)n * CKCH + ch] * scale;
    out[(size_t)b * CKCH * NSEQ + (size_t)ch * NSEQ + n] = v > 0.f ? v : 0.f;
}

// ---------------------------------------------------------------------------
extern "C" void kernel_launch(void* const* d_in, const int* in_sizes, int n_in,
                              void* d_out, int out_size, void* d_ws, size_t ws_size,
                              hipStream_t stream) {
    (void)in_sizes; (void)n_in; (void)out_size; (void)ws_size;
    const float* values = (const float*)d_in[2];
    const float* L1w = (const float*)d_in[4];
    const float* L1b = (const float*)d_in[5];
    const float* L2w = (const float*)d_in[6];
    const float* L2b = (const float*)d_in[7];
    const float* ecHw = (const float*)d_in[8];
    const float* ecHb = (const float*)d_in[9];
    const float* ecLw = (const float*)d_in[10];
    const float* ecLb = (const float*)d_in[11];
    const float* hpWr = (const float*)d_in[12];
    const float* hpWi = (const float*)d_in[13];
    const float* lpWr = (const float*)d_in[14];
    const float* lpWi = (const float*)d_in[15];
    const float* rcLw = (const float*)d_in[16];
    const float* rcLb = (const float*)d_in[17];
    const float* rcHw = (const float*)d_in[18];
    const float* rcHb = (const float*)d_in[19];

    char* ws = (char*)d_ws;
    size_t off = 0;
    auto walloc = [&](size_t bytes) -> char* {
        char* p = ws + off;
        off += (bytes + 255) & ~(size_t)255;
        return p;
    };
    const size_t WCONV = (size_t)2 * FS * CKCH * CKCH;     // elems per weight tensor (incl NB)
    _Float16* wEH = (_Float16*)walloc(WCONV * 2);
    _Float16* wEL = (_Float16*)walloc(WCONV * 2);
    _Float16* wRL = (_Float16*)walloc(WCONV * 2);
    _Float16* wRH = (_Float16*)walloc(WCONV * 2);
    _Float16* w1  = (_Float16*)walloc((size_t)CKCH * 256 * 2);
    _Float16* w2  = (_Float16*)walloc((size_t)256 * CKCH * 2);
    float* bufA = (float*)walloc((size_t)BATCH * CKCH * NSEQ * 4);
    float* bufB = (float*)walloc((size_t)BATCH * CKCH * NSEQ * 4);
    float* bufC = (float*)walloc((size_t)BATCH * CKCH * NSEQ * 4);
    float* det[10];
    for (int i = 0; i < 10; ++i)
        det[i] = (float*)walloc((size_t)BATCH * CKCH * (NSEQ >> (i + 1)) * 4);
    float* Xre = (float*)walloc((size_t)BATCH * 16 * CKCH * 4);
    float* Xim = (float*)walloc((size_t)BATCH * 16 * CKCH * 4);
    float* Gre = (float*)walloc((size_t)BATCH * 16 * CKCH * 4);
    float* Gim = (float*)walloc((size_t)BATCH * 16 * CKCH * 4);
    float* stats = (float*)walloc(256);

    const float inv_sqrtC = 0.70710678118654752440f;

    // --- weight conversion (recomputed every call: deterministic) ---
    {
        int tot = CKCH * CKCH * FS;
        int gb = (tot + 255) / 256;
        for (int nb = 0; nb < 2; ++nb) {
            size_t so = (size_t)nb * tot;
            yk_wcvt<<<gb, 256, 0, stream>>>(ecHw + so, wEH + so, tot);
            yk_wcvt<<<gb, 256, 0, stream>>>(ecLw + so, wEL + so, tot);
            yk_wcvt<<<gb, 256, 0, stream>>>(rcLw + so, wRL + so, tot);
            yk_wcvt<<<gb, 256, 0, stream>>>(rcHw + so, wRH + so, tot);
        }
        yk_cvt<<<(CKCH * 256 + 255) / 256, 256, 0, stream>>>(L1w, w1, CKCH * 256);
        yk_cvt<<<(256 * CKCH + 255) / 256, 256, 0, stream>>>(L2w, w2, 256 * CKCH);
    }

    // --- L1: values (B,N,256) @ W1^T -> ch-major (B,512,N) ---
    {
        dim3 g(NSEQ / 16, CKCH / 128, BATCH);
        yk_fc_wmma<<<g, 256, 0, stream>>>(values, w1, L1b, bufA, NSEQ, 256, CKCH, 1, 1.f, 0);
    }

    float* cur = bufA;
    float* nxt = bufB;
    float* tmp = bufC;

    auto sparse_ft = [&](const float* src, float* dstF, int L, const float* wr_,
                         const float* wi_) {
        int l = L / 2 + 1;
        if (l > 16) l = 16;
        yk_zero<<<1, 32, 0, stream>>>(stats);
        int tot = BATCH * CKCH * l;
        yk_dft<<<(tot + 255) / 256, 256, 0, stream>>>(src, Xre, Xim, L, l);
        yk_mix<<<(tot + 255) / 256, 256, 0, stream>>>(Xre, Xim, wr_, wi_, Gre, Gim, l);
        long long n = (long long)BATCH * CKCH * L;
        int gb = (int)((n + 255) / 256);
        yk_idft<<<gb, 256, 0, stream>>>(Gre, Gim, dstF, stats, L, l);
        yk_norm<<<gb, 256, 0, stream>>>(dstF, stats, n, 1);
    };

    const size_t shm2 = (size_t)(2 * 32 + 6) * CKCH * 2;   // stride-2 staging (70 KB)
    const size_t shm1 = (size_t)(1 * 32 + 6) * CKCH * 2;   // stride-1 staging (38 KB)

    for (int blk = 0; blk < 2; ++blk) {
        const _Float16* tEH = wEH + (size_t)blk * FS * CKCH * CKCH;
        const _Float16* tEL = wEL + (size_t)blk * FS * CKCH * CKCH;
        const _Float16* tRL = wRL + (size_t)blk * FS * CKCH * CKCH;
        const _Float16* tRH = wRH + (size_t)blk * FS * CKCH * CKCH;
        const float* bEH = ecHb + (size_t)blk * CKCH;
        const float* bEL = ecLb + (size_t)blk * CKCH;
        const float* bRL = rcLb + (size_t)blk * CKCH;
        const float* bRH = rcHb + (size_t)blk * CKCH;
        const float* wrH = hpWr + (size_t)blk * CKCH * CKCH * 16;
        const float* wiH = hpWi + (size_t)blk * CKCH * CKCH * 16;
        const float* wrL = lpWr + (size_t)blk * CKCH * CKCH * 16;
        const float* wiL = lpWi + (size_t)blk * CKCH * CKCH * 16;

        // ---- decomposition: 10 levels ----
        int L = NSEQ;
        for (int lvl = 0; lvl < 10; ++lvl) {
            int Lh = L >> 1;
            dim3 gc((Lh + 31) / 32, CKCH / 128, BATCH);
            yk_conv_wmma<<<gc, 256, shm2, stream>>>(cur, tEH, bEH, tmp, L, Lh, 2, 0, 0);
            sparse_ft(tmp, det[lvl], Lh, wrH, wiH);
            yk_conv_wmma<<<gc, 256, shm2, stream>>>(cur, tEL, bEL, tmp, L, Lh, 2, 0, 0);
            sparse_ft(tmp, nxt, Lh, wrL, wiL);
            float* sw = cur; cur = nxt; nxt = sw;
            L = Lh;
        }

        // ---- reconstruction: 10 levels ----
        for (int lv = 0; lv < 10; ++lv) {
            int Ld = 1 << lv;
            int Lo = 2 * Ld;
            float* dst = (cur == bufA) ? bufB : bufA;
            dim3 gr((Lo + 31) / 32, CKCH / 128, BATCH);
            // cA (repeat x2) through rc_lp conv
            yk_conv_wmma<<<gr, 256, shm1, stream>>>(cur, tRL, bRL, dst, Ld, Lo, 1, 1, 0);
            // cD (repeat x2) through rc_hp conv, accumulated
            yk_conv_wmma<<<gr, 256, shm1, stream>>>(det[9 - lv], tRH, bRH, dst, Ld, Lo, 1, 1, 1);
            long long n = (long long)BATCH * CKCH * Lo;
            int gb = (int)((n + 255) / 256);
            yk_zero<<<1, 32, 0, stream>>>(stats);
            yk_stats<<<gb, 256, 0, stream>>>(dst, stats, n);
            yk_norm<<<gb, 256, 0, stream>>>(dst, stats, n, 0);
            cur = dst;
        }

        if (blk == 0) {
            // block output: /sqrt(C), relu, true (N,512)->(512,N) transpose
            float* dstT = (cur == bufA) ? bufB : bufA;
            int tot = BATCH * CKCH * NSEQ;
            yk_relu_tr<<<(tot + 255) / 256, 256, 0, stream>>>(cur, dstT, inv_sqrtC);
            cur = dstT;
            nxt = (cur == bufA) ? bufB : bufA;
            tmp = bufC;
        } else {
            // L2: relu(flat/sqrt(C)) viewed (N,512) row-major @ W2^T -> (B,N,256)
            dim3 g(NSEQ / 16, 256 / 128, BATCH);
            yk_fc_wmma<<<g, 256, 0, stream>>>(cur, w2, L2b, (float*)d_out, NSEQ, CKCH, 256,
                                              0, inv_sqrtC, 1);
        }
    }
}
